// SearchPredictModel_40621800685578
// MI455X (gfx1250) — compile-verified, matrix operand order checked by reading, not run on hardware
//
#include <hip/hip_runtime.h>
#include <hip/hip_bf16.h>
#include <math.h>

#define PAD_TOK 200000

typedef __attribute__((ext_vector_type(16))) _Float16 v16h;
typedef __attribute__((ext_vector_type(8)))  float    v8f;

union FragH { v16h v; unsigned int u[8]; };
union FragF { v8f v; float f[8]; };
union HalfPair { _Float16 h[2]; unsigned int u; };

__device__ __forceinline__ int kpair_of(int v, int hi) {
    int base = (v < 4) ? (2 * v) : (16 + 2 * (v - 4));
    return base + (hi ? 8 : 0);
}

// Load a 16x32 (A) / 32x16 (B) f16 fragment from a row-major [16][ld] tile.
// Lane L: row = L%16; VGPR v holds the K-pair kpair_of(v, L>=16) (AMD table order).
__device__ __forceinline__ v16h load_frag_h(const _Float16* base, int ld) {
    int lane = threadIdx.x & 31;
    int m = lane & 15, hi = (lane >> 4) & 1;
    FragH f;
#pragma unroll
    for (int v = 0; v < 8; ++v)
        f.u[v] = *(const unsigned int*)(base + m * ld + kpair_of(v, hi));
    return f.v;
}

// Same fragment, but source is f32 (convert to f16 in registers).
__device__ __forceinline__ v16h load_fragA_f32(const float* base, int ld) {
    int lane = threadIdx.x & 31;
    int m = lane & 15, hi = (lane >> 4) & 1;
    FragH f;
#pragma unroll
    for (int v = 0; v < 8; ++v) {
        int k = kpair_of(v, hi);
        float2 d = *(const float2*)(base + m * ld + k);
        HalfPair p; p.h[0] = (_Float16)d.x; p.h[1] = (_Float16)d.y;
        f.u[v] = p.u;
    }
    return f.v;
}

__device__ __forceinline__ v8f wmma16(v16h a, v16h b, v8f c) {
    return __builtin_amdgcn_wmma_f32_16x16x32_f16(false, a, false, b, (short)0, c,
                                                  false, false);
}

__device__ __forceinline__ float sigm(float x) { return 1.f / (1.f + __expf(-x)); }

// ---------------------------------------------------------------------------
// Weight prep: dst[n][k] = f16( k<K1 ? src1[k*N+n] : src2[(k-K1)*N+n] )
// ---------------------------------------------------------------------------
__global__ void prep_kernel(const float* __restrict__ src1,
                            const float* __restrict__ src2,
                            _Float16* __restrict__ dst,
                            int K1, int K, int N) {
    int i = blockIdx.x * blockDim.x + threadIdx.x;
    if (i >= N * K) return;
    int n = i / K, k = i - n * K;
    float v = (k < K1) ? src1[k * N + n] : src2[(k - K1) * N + n];
    dst[i] = (_Float16)v;
}

// ---------------------------------------------------------------------------
// x_self[b][d] = sum_{j<6} emb[x[b,0,self_loc[b],j]][d]
// ---------------------------------------------------------------------------
__global__ void xself_kernel(const int* __restrict__ x,
                             const int* __restrict__ sloc,
                             const float* __restrict__ emb,
                             float* __restrict__ xself) {
    int b = blockIdx.x, d = threadIdx.x;     // 128 threads
    const int* xi = x + ((size_t)(b * 16) * 64 + sloc[b]) * 8;
    float s = 0.f;
#pragma unroll
    for (int j = 0; j < 6; ++j) s += emb[(size_t)xi[j] * 128 + d];
    xself[b * 128 + d] = s;
}

// ---------------------------------------------------------------------------
// Per (b,s): cosine sim of e_self vs 64 items, top-15 (lowest index wins
// ties -> matches reference's global-min "neg" masking), sorted indices,
// self_loc override on slot 14 at s==0, then gather tokens + dt.
// ---------------------------------------------------------------------------
__global__ void topk_kernel(const int* __restrict__ x,
                            const int* __restrict__ sloc,
                            const float* __restrict__ emb,
                            int* __restrict__ xt_tok,
                            float* __restrict__ xt_dt) {
    int b = blockIdx.x >> 4, s = blockIdx.x & 15;
    int tid = threadIdx.x;                   // 64 threads
    __shared__ float es[128];
    __shared__ float vals[64];
    __shared__ int sel[15];

    int stok = x[((size_t)(b * 16) * 64 + sloc[b]) * 8 + 5];
    for (int d = tid; d < 128; d += 64) es[d] = emb[(size_t)stok * 128 + d];
    __syncthreads();

    {
        const int* xi = x + ((size_t)(b * 16 + s) * 64 + tid) * 8;
        int tok = xi[5];
        const float* er = emb + (size_t)tok * 128;
        float dot = 0.f, na = 0.f, nb = 0.f;
        for (int d = 0; d < 128; ++d) {
            float a = es[d], bb = er[d];
            dot += a * bb; na += a * a; nb += bb * bb;
        }
        float cs = dot / (fmaxf(sqrtf(na), 1e-8f) * fmaxf(sqrtf(nb), 1e-8f));
        vals[tid] = (xi[0] != PAD_TOK) ? cs : -3.0e38f;
    }
    __syncthreads();

    if (tid == 0) {
        unsigned long long used = 0ull;
        for (int k = 0; k < 15; ++k) {
            float best = 0.f; int bi = -1;
            for (int i = 0; i < 64; ++i)
                if (!((used >> i) & 1ull) && (bi < 0 || vals[i] > best)) {
                    best = vals[i]; bi = i;
                }
            used |= 1ull << bi;
            sel[k] = bi;
        }
        for (int k = 1; k < 15; ++k) {        // sort ascending
            int v = sel[k], j = k - 1;
            while (j >= 0 && sel[j] > v) { sel[j + 1] = sel[j]; --j; }
            sel[j + 1] = v;
        }
        if (s == 0) sel[14] = sloc[b];
    }
    __syncthreads();

    int r = b * 16 + s;
    if (tid < 15) {
        int k = tid;
        const int* xi = x + ((size_t)(b * 16 + s) * 64 + sel[k]) * 8;
#pragma unroll
        for (int j = 0; j < 6; ++j) xt_tok[((size_t)r * 15 + k) * 6 + j] = xi[j];
        float dt = 0.f;
        if (k > 0) {
            const int* xp = x + ((size_t)(b * 16 + s) * 64 + sel[k - 1]) * 8;
            dt = (float)xi[6] - (float)xp[6];
        }
        xt_dt[(size_t)r * 15 + k] = dt;
    }
}

// ---------------------------------------------------------------------------
// Time-decay GRU. One wave handles 16 rows, h kept in LDS (f32).
// A-stage (f16): [ f (F cols) | hd (256 cols) ], WT (f16) is [768][F+256]
// with rows 0..255 = z-gate cols, 256..511 = r, 512..767 = n (W part in
// k<F, U part in k>=F).  Each A fragment feeds 3 WMMAs (z/r + g2 or u2).
// MODE 0: xs RNN (F=256, f = [x_self|sum-emb]); MODE 1: continuous (F=128).
// ---------------------------------------------------------------------------
template<int F, int MODE>
__global__ void rnn_kernel(const _Float16* __restrict__ WT,
                           const float* __restrict__ bias,
                           const float* __restrict__ wdec,
                           const float* __restrict__ xself,
                           const int* __restrict__ xt_tok,
                           const float* __restrict__ xt_dt,
                           const int* __restrict__ xc,
                           const float* __restrict__ emb,
                           float* __restrict__ hout,
                           int T) {
    constexpr int KT = F + 256;
    extern __shared__ char smem_raw[];
    _Float16* Ast = (_Float16*)smem_raw;          // [16][KT]
    __shared__ float hsh[16 * 256];
    __shared__ float dts[16];
    const int lane = threadIdx.x;                 // 32 threads = 1 wave
    const int nlo = lane & 15, hi = (lane >> 4) & 1;
    const int row0 = blockIdx.x * 16;

    for (int i = lane; i < 16 * 256; i += 32) hsh[i] = 0.f;
    if (MODE == 0) {                               // static x_self feature part
        for (int i = lane; i < 16 * 128; i += 32) {
            int r = i >> 7, d = i & 127;
            int b = (row0 + r) >> 4;
            Ast[r * KT + d] = (_Float16)xself[b * 128 + d];
        }
    }
    __syncthreads();

    for (int t = 0; t < T; ++t) {
        if (lane < 16) {
            int gr = row0 + lane;
            float dt = 0.f;
            if (MODE == 0) dt = xt_dt[(size_t)gr * 15 + t];
            else if (t > 0)
                dt = (float)xc[((size_t)gr * 128 + t) * 8 + 6] -
                     (float)xc[((size_t)gr * 128 + t - 1) * 8 + 6];
            dts[lane] = dt;
        }
        const int foff = (MODE == 0) ? 128 : 0;
        for (int i = lane; i < 16 * 128; i += 32) {   // per-step embedding sum
            int r = i >> 7, d = i & 127, gr = row0 + r;
            const int* tk = (MODE == 0) ? (xt_tok + ((size_t)gr * 15 + t) * 6)
                                        : (xc + ((size_t)gr * 128 + t) * 8);
            float s = 0.f;
#pragma unroll
            for (int j = 0; j < 6; ++j) s += emb[(size_t)tk[j] * 128 + d];
            Ast[r * KT + foff + d] = (_Float16)s;
        }
        __syncthreads();
        // hd = h * exp(-relu(dt*wt)) -> f16 A-stage
        for (int c = 0; c < 16; ++c) {
            int col = c * 16 + nlo;
            float w = wdec[col];
#pragma unroll
            for (int v = 0; v < 8; ++v) {
                int m = v + 8 * hi;
                float dec = __expf(-fmaxf(dts[m] * w, 0.f));
                Ast[m * KT + F + col] = (_Float16)(hsh[m * 256 + col] * dec);
            }
        }
        __syncthreads();
        // gates per 16-column chunk; A fragments reused across 3 gate GEMMs
        for (int c = 0; c < 16; ++c) {
            const _Float16* Bz = WT + (size_t)(c * 16) * KT;
            const _Float16* Br = WT + (size_t)(256 + c * 16) * KT;
            const _Float16* Bn = WT + (size_t)(512 + c * 16) * KT;
            if (c < 15)                               // warm L0/L2 for next chunk
                __builtin_prefetch(WT + (size_t)((c + 1) * 16) * KT, 0, 1);
            v8f az = {}, ar = {}, ag = {}, au = {};
            for (int kb = 0; kb < F; kb += 32) {      // k < F: z, r, g2 (f part)
                v16h a = load_frag_h(Ast + kb, KT);
                az = wmma16(a, load_frag_h(Bz + kb, KT), az);
                ar = wmma16(a, load_frag_h(Br + kb, KT), ar);
                ag = wmma16(a, load_frag_h(Bn + kb, KT), ag);
            }
            for (int kb = F; kb < KT; kb += 32) {     // k >= F: z, r, u2 (hd part)
                v16h a = load_frag_h(Ast + kb, KT);
                az = wmma16(a, load_frag_h(Bz + kb, KT), az);
                ar = wmma16(a, load_frag_h(Br + kb, KT), ar);
                au = wmma16(a, load_frag_h(Bn + kb, KT), au);
            }
            int col = c * 16 + nlo;
            float bz = bias[col], br = bias[256 + col], bn = bias[512 + col];
            FragF fz, fr, fg, fu;
            fz.v = az; fr.v = ar; fg.v = ag; fu.v = au;
#pragma unroll
            for (int v = 0; v < 8; ++v) {
                int m = v + 8 * hi;
                float z = sigm(fz.f[v] + bz);
                float r = sigm(fr.f[v] + br);
                float n = tanhf(fg.f[v] + bn + r * fu.f[v]);
                float hd = (float)Ast[m * KT + F + col];
                hsh[m * 256 + col] = (1.f - z) * hd + z * n;
            }
        }
        __syncthreads();
        if (MODE == 1 && t >= T - 4) {                // stream last 4 steps
            int j = t - (T - 4);
            for (int i = lane; i < 16 * 256; i += 32) {
                int r = i >> 8, col = i & 255;
                hout[((size_t)(row0 + r) * 4 + j) * 256 + col] = hsh[i];
            }
            __syncthreads();
        }
    }
    if (MODE == 0) {
        for (int i = lane; i < 16 * 256; i += 32) {
            int r = i >> 8, col = i & 255;
            hout[(size_t)(row0 + r) * 256 + col] = hsh[i];
        }
    }
}

// ---------------------------------------------------------------------------
// Attention + hh assembly: hh = [x_self | h[:,0] | h_att(M*H) | h_cont]
// ---------------------------------------------------------------------------
__global__ void att_kernel(const float* __restrict__ hout,
                           const float* __restrict__ Wv,
                           const float* __restrict__ bv,
                           const float* __restrict__ xself,
                           const float* __restrict__ hcont,
                           float* __restrict__ hh) {
    int b = blockIdx.x;
    __shared__ float hs[16 * 256];
    __shared__ float a2[16 * 4];
    int tid = threadIdx.x;                        // 256 threads
    for (int i = tid; i < 16 * 256; i += 256) hs[i] = hout[(size_t)(b * 16) * 256 + i];
    __syncthreads();
    if (tid < 64) {
        int s = tid >> 2, m = tid & 3;
        float acc = bv[m];
        for (int k = 0; k < 256; ++k) acc += hs[s * 256 + k] * Wv[k * 4 + m];
        for (int k = 0; k < 256; ++k) acc += hs[k] * Wv[(256 + k) * 4 + m];
        a2[s * 4 + m] = tanhf(acc);
    }
    __syncthreads();
    if (tid < 4) {                                 // softmax over s
        int m = tid;
        float mx = -1e30f;
        for (int s = 0; s < 16; ++s) mx = fmaxf(mx, a2[s * 4 + m]);
        float e[16], sum = 0.f;
        for (int s = 0; s < 16; ++s) { e[s] = __expf(a2[s * 4 + m] - mx); sum += e[s]; }
        for (int s = 0; s < 16; ++s) a2[s * 4 + m] = e[s] / sum;
    }
    __syncthreads();
    float* row = hh + (size_t)b * 1664;
    for (int d = tid; d < 128; d += 256) row[d] = xself[b * 128 + d];
    for (int d = tid; d < 256; d += 256) row[128 + d] = hs[d];
    for (int i = tid; i < 4 * 256; i += 256) {
        int m = i >> 8, hc = i & 255;
        float acc = 0.f;
        for (int s = 0; s < 16; ++s) acc += a2[s * 4 + m] * hs[s * 256 + hc];
        row[384 + i] = acc;
    }
    for (int d = tid; d < 256; d += 256) row[1408 + d] = hcont[(size_t)b * 256 + d];
}

// ---------------------------------------------------------------------------
// WMMA GEMM + optional leaky-relu: out = act(A[512][Kc] @ WT[Nc][Kc]^T + b)
// One wave per 16x16 output tile.
// ---------------------------------------------------------------------------
__global__ void gemm_act_kernel(const float* __restrict__ A,
                                const _Float16* __restrict__ WT,
                                const float* __restrict__ bias,
                                float* __restrict__ out,
                                int Kc, int Nc, int act) {
    int gw = (blockIdx.x * blockDim.x + threadIdx.x) >> 5;
    int ntiles = Nc >> 4;
    int mt = gw / ntiles, nt = gw - mt * ntiles;
    int lane = threadIdx.x & 31, nlo = lane & 15, hi = (lane >> 4) & 1;
    const float* Ab = A + (size_t)(mt * 16) * Kc;
    const _Float16* Bb = WT + (size_t)(nt * 16) * Kc;
    v8f acc = {};
    for (int kb = 0; kb < Kc; kb += 32)
        acc = wmma16(load_fragA_f32(Ab + kb, Kc), load_frag_h(Bb + kb, Kc), acc);
    float bv = bias[nt * 16 + nlo];
    FragF f; f.v = acc;
#pragma unroll
    for (int v = 0; v < 8; ++v) {
        float x = f.f[v] + bv;
        if (act) x = (x > 0.f) ? x : 0.01f * x;
        out[(size_t)(mt * 16 + v + 8 * hi) * Nc + nt * 16 + nlo] = x;
    }
}

__global__ void final_kernel(const float* __restrict__ z2,
                             const float* __restrict__ Wp,
                             const float* __restrict__ bp,
                             float* __restrict__ out) {
    int b = blockIdx.x * blockDim.x + threadIdx.x;
    if (b >= 512) return;
    float acc = bp[0];
    for (int k = 0; k < 256; ++k) acc += z2[(size_t)b * 256 + k] * Wp[k];
    out[b] = 1.f / (1.f + __expf(-acc));
}

// ---------------------------------------------------------------------------
extern "C" void kernel_launch(void* const* d_in, const int* in_sizes, int n_in,
                              void* d_out, int out_size, void* d_ws, size_t ws_size,
                              hipStream_t stream) {
    const int*   x      = (const int*)d_in[0];
    const int*   xc     = (const int*)d_in[1];
    const int*   sloc   = (const int*)d_in[2];
    const float* emb    = (const float*)d_in[3];
    const float* W_rnn  = (const float*)d_in[4];
    const float* U_rnn  = (const float*)d_in[5];
    const float* b_rnn  = (const float*)d_in[6];
    const float* wt_rnn = (const float*)d_in[7];
    const float* W_c    = (const float*)d_in[8];
    const float* U_c    = (const float*)d_in[9];
    const float* b_c    = (const float*)d_in[10];
    const float* wt_c   = (const float*)d_in[11];
    const float* Wv     = (const float*)d_in[12];
    const float* bv     = (const float*)d_in[13];
    const float* W0     = (const float*)d_in[14];
    const float* b0     = (const float*)d_in[15];
    const float* W1     = (const float*)d_in[16];
    const float* b1     = (const float*)d_in[17];
    const float* W2     = (const float*)d_in[18];
    const float* b2     = (const float*)d_in[19];
    const float* Wp     = (const float*)d_in[20];
    const float* bp     = (const float*)d_in[21];
    float* out = (float*)d_out;

    char* ws = (char*)d_ws;
    size_t off = 0;
    auto take = [&](size_t bytes) {
        char* p = ws + off;
        off += (bytes + 255) & ~(size_t)255;
        return p;
    };
    _Float16* WT_rnn = (_Float16*)take(768ull * 512 * 2);
    _Float16* WT_c   = (_Float16*)take(768ull * 384 * 2);
    _Float16* W0T    = (_Float16*)take(1024ull * 1664 * 2);
    _Float16* W1T    = (_Float16*)take(512ull * 1024 * 2);
    _Float16* W2T    = (_Float16*)take(256ull * 512 * 2);
    float* xself  = (float*)take(512ull * 128 * 4);
    int*   xt_tok = (int*)take(8192ull * 15 * 6 * 4);
    float* xt_dt  = (float*)take(8192ull * 15 * 4);
    float* hout   = (float*)take(8192ull * 256 * 4);
    float* hcont  = (float*)take(512ull * 256 * 4);
    float* hh     = (float*)take(512ull * 1664 * 4);
    float* z0     = (float*)take(512ull * 1024 * 4);
    float* z1     = (float*)take(512ull * 512 * 4);
    float* z2     = (float*)take(512ull * 256 * 4);

    auto cdiv = [](int a, int b) { return (a + b - 1) / b; };

    prep_kernel<<<cdiv(768 * 512, 256), 256, 0, stream>>>(W_rnn, U_rnn, WT_rnn, 256, 512, 768);
    prep_kernel<<<cdiv(768 * 384, 256), 256, 0, stream>>>(W_c, U_c, WT_c, 128, 384, 768);
    prep_kernel<<<cdiv(1024 * 1664, 256), 256, 0, stream>>>(W0, W0, W0T, 1664, 1664, 1024);
    prep_kernel<<<cdiv(512 * 1024, 256), 256, 0, stream>>>(W1, W1, W1T, 1024, 1024, 512);
    prep_kernel<<<cdiv(256 * 512, 256), 256, 0, stream>>>(W2, W2, W2T, 512, 512, 256);

    xself_kernel<<<512, 128, 0, stream>>>(x, sloc, emb, xself);
    topk_kernel<<<8192, 64, 0, stream>>>(x, sloc, emb, xt_tok, xt_dt);

    rnn_kernel<256, 0><<<512, 32, 16 * 512 * 2, stream>>>(
        WT_rnn, b_rnn, wt_rnn, xself, xt_tok, xt_dt, nullptr, emb, hout, 15);
    rnn_kernel<128, 1><<<8, 32, 16 * 384 * 2, stream>>>(
        WT_c, b_c, wt_c, nullptr, nullptr, nullptr, xc, emb, hcont, 128);

    att_kernel<<<512, 256, 0, stream>>>(hout, Wv, bv, xself, hcont, hh);

    gemm_act_kernel<<<512, 128, 0, stream>>>(hh, W0T, b0, z0, 1664, 1024, 1);
    gemm_act_kernel<<<256, 128, 0, stream>>>(z0, W1T, b1, z1, 1024, 512, 1);
    gemm_act_kernel<<<128, 128, 0, stream>>>(z1, W2T, b2, z2, 512, 256, 1);
    final_kernel<<<2, 256, 0, stream>>>(z2, Wp, bp, out);
}
